// LSHAttention_56100862820695
// MI455X (gfx1250) — matrix operand order repaired and checked by reading, not compiled, not run on hardware
//
#include <hip/hip_runtime.h>

// ---------------- problem constants ----------------
#define BATCH 4
#define SEQ   8192
#define DIM   1024
#define HEADS 8
#define DH    128          // DIM / HEADS
#define NBUCK 128          // SEQ / 64
#define BUCK  64
#define TOK   (BATCH * SEQ)   // 32768

typedef __attribute__((ext_vector_type(16))) __bf16 v16bf;
typedef __attribute__((ext_vector_type(8)))  __bf16 v8bf;
typedef __attribute__((ext_vector_type(8)))  float  v8f;

// ---------------- gfx1250 async global->LDS copy (guarded) ----------------
#if __has_builtin(__builtin_amdgcn_global_load_async_to_lds_b128) && \
    __has_builtin(__builtin_amdgcn_s_wait_asynccnt)
#define HAS_ASYNC_LDS 1
#else
#define HAS_ASYNC_LDS 0
#endif

// exact parameter types per hipcc diagnostic: pointer to int __vector(4),
// global (AS1) source and LDS (AS3) destination
typedef int v4i_vec __attribute__((vector_size(16)));
typedef v4i_vec __attribute__((address_space(1))) as1_v4i;
typedef v4i_vec __attribute__((address_space(3))) as3_v4i;

// copy 8 bf16 (16 bytes) from global to LDS, per lane
__device__ __forceinline__ void copy_b128(const __bf16* g, __bf16* l) {
#if HAS_ASYNC_LDS
  __builtin_amdgcn_global_load_async_to_lds_b128(
      (as1_v4i*)(g), (as3_v4i*)(l), 0, 0);
#else
  *(uint4*)l = *(const uint4*)g;
#endif
}
__device__ __forceinline__ void async_fence() {
#if HAS_ASYNC_LDS
  __builtin_amdgcn_s_wait_asynccnt(0);
#endif
}

// ---------------- WMMA helpers (CDNA5 wave32 layouts) ----------------
__device__ __forceinline__ v8f wmma_bf16(v16bf a, v16bf b, v8f c) {
  return __builtin_amdgcn_wmma_f32_16x16x32_bf16(
      false, a, false, b, (short)0, c, false, false);
}

// A fragment from row-major bf16 LDS matrix [row][stride].
// lanes 0-15: M=lane, K in {0..15}; lanes 16-31: M=lane-16, K in {16..31}
__device__ __forceinline__ v16bf load_a_frag(const __bf16* lds, int stride,
                                             int m_base, int k_base, int lane) {
  int m  = m_base + (lane & 15);
  int hi = (lane >> 4) & 1;
  const __bf16* row = lds + m * stride + k_base + hi * 16;
  v16bf a;
  *((uint4*)&a)     = *(const uint4*)(row);
  *((uint4*)&a + 1) = *(const uint4*)(row + 8);
  return a;
}

// B fragment where B is stored TRANSPOSED in LDS as Bt[n][k]:
// lane gives N (=lane&15), K = i + 16*hi -> two contiguous 16B reads.
__device__ __forceinline__ v16bf load_bt_frag(const __bf16* lds, int stride,
                                              int n_base, int k_base, int lane) {
  int n  = n_base + (lane & 15);
  int hi = (lane >> 4) & 1;
  const __bf16* p = lds + n * stride + k_base + hi * 16;
  v16bf b;
  *((uint4*)&b)     = *(const uint4*)(p);
  *((uint4*)&b + 1) = *(const uint4*)(p + 8);
  return b;
}

// ---------------- small utility kernels ----------------
__global__ __launch_bounds__(256) void convert_bf16_kernel(const float* __restrict__ in,
                                                           __bf16* __restrict__ out, int n) {
  int i = blockIdx.x * 256 + threadIdx.x;
  if (i < n) out[i] = (__bf16)in[i];
}

// One block per token: bf16 copy of the x row + per-head LSH angle.
__global__ __launch_bounds__(256) void hash_convert_kernel(const float* __restrict__ x,
                                                           const float* __restrict__ Whash,
                                                           float* __restrict__ angles,
                                                           __bf16* __restrict__ xb) {
  __shared__ float xrow[DIM];
  int tokId = blockIdx.x;
  const float* src = x + (size_t)tokId * DIM;
  int t = threadIdx.x;
#pragma unroll
  for (int j = 0; j < 4; ++j) {
    float v = src[t * 4 + j];
    xrow[t * 4 + j] = v;
    xb[(size_t)tokId * DIM + t * 4 + j] = (__bf16)v;
  }
  __syncthreads();

  int wave = t >> 5;  // wave == head
  int lane = t & 31;
  float a0 = 0.f, a1 = 0.f;
  for (int d = lane; d < DIM; d += 32) {
    float xv = xrow[d];
    a0 += xv * Whash[d * (2 * HEADS) + 2 * wave];
    a1 += xv * Whash[d * (2 * HEADS) + 2 * wave + 1];
  }
#pragma unroll
  for (int off = 16; off > 0; off >>= 1) {
    a0 += __shfl_xor(a0, off, 32);
    a1 += __shfl_xor(a1, off, 32);
  }
  if (lane == 0) {
    int b = tokId / SEQ, s = tokId % SEQ;
    angles[((size_t)b * HEADS + wave) * SEQ + s] = a0 / (a1 + 1e-4f);
  }
}

// One block per (b,h): bitonic argsort of 8192 angle keys (ascending).
__global__ __launch_bounds__(1024) void bitonic_sort_kernel(const float* __restrict__ angles,
                                                            int* __restrict__ idx) {
  __shared__ float          key[SEQ];   // 32 KB
  __shared__ unsigned short val[SEQ];   // 16 KB
  size_t base = (size_t)blockIdx.x * SEQ;
  int t = threadIdx.x;
  for (int i = t; i < SEQ; i += 1024) { key[i] = angles[base + i]; val[i] = (unsigned short)i; }
  __syncthreads();
  for (int k = 2; k <= SEQ; k <<= 1) {
    for (int j = k >> 1; j > 0; j >>= 1) {
      for (int p = t; p < SEQ / 2; p += 1024) {
        int i  = 2 * p - (p & (j - 1));
        int ix = i | j;
        bool up = ((i & k) == 0);
        float ki = key[i], kx = key[ix];
        if ((ki > kx) == up) {
          key[i] = kx; key[ix] = ki;
          unsigned short vi = val[i]; val[i] = val[ix]; val[ix] = vi;
        }
      }
      __syncthreads();
    }
  }
  for (int i = t; i < SEQ; i += 1024) idx[base + i] = (int)val[i];
}

// ---------------- fused multi-output WMMA GEMM -----------------------------
// C_m[M,N] = A(bf16,[M,K]) @ W_m(bf16,[K][N]) + bias_m    for m in [0,NMAT)
// BM=128, BN=64, BK=32; 256 threads = 8 waves, each wave 2x2 tiles per matrix.
// Double-buffered LDS; A staged with async global->LDS when available.
template <int NMAT, typename OutT>
__global__ __launch_bounds__(256) void gemm_multi_kernel(
    const __bf16* __restrict__ A,
    const __bf16* __restrict__ W0, const __bf16* __restrict__ W1,
    const float* __restrict__ bias0, const float* __restrict__ bias1,
    OutT* __restrict__ C0, OutT* __restrict__ C1,
    int M, int N, int K) {
  __shared__ __align__(16) __bf16 As[2][128 * 48];        // [m][k], stride 48
  __shared__ __align__(16) __bf16 Wt[2][NMAT][64 * 40];   // transposed [n][k], stride 40

  int n0 = blockIdx.x * 64;
  int m0 = blockIdx.y * 128;
  int t  = threadIdx.x;
  int lane = t & 31, wave = t >> 5;
  int wm = (wave >> 1) * 32;
  int wn = (wave & 1) * 32;
  int hi = (lane >> 4) & 1;

  const __bf16* Ws[2] = {W0, W1};
  const float*  bs[2] = {bias0, bias1};
  OutT*         Cs[2] = {C0, C1};

  v8f acc[NMAT][2][2];
#pragma unroll
  for (int m = 0; m < NMAT; ++m)
#pragma unroll
    for (int i = 0; i < 2; ++i)
#pragma unroll
      for (int j = 0; j < 2; ++j) acc[m][i][j] = (v8f)0.0f;

  // tile staging: A via (async) b128 copies, W transposed via scalar stores
  auto stage = [&](int buf, int k0) {
    {
      int row = t >> 1, seg = t & 1;
      const __bf16* g = A + (size_t)(m0 + row) * K + k0 + seg * 16;
      __bf16* l = &As[buf][row * 48 + seg * 16];
      copy_b128(g, l);
      copy_b128(g + 8, l + 8);
    }
    int kr = t >> 3, nc = (t & 7) * 8;
#pragma unroll
    for (int m = 0; m < NMAT; ++m) {
      const __bf16* gw = Ws[m] + (size_t)(k0 + kr) * N + n0 + nc;
      if (k0 + 32 < K) __builtin_prefetch(gw + (size_t)32 * N, 0, 0);
      v8bf wv = *(const v8bf*)gw;
#pragma unroll
      for (int j = 0; j < 8; ++j) Wt[buf][m][(nc + j) * 40 + kr] = wv[j];
    }
  };

  auto compute = [&](int buf) {
    v16bf a0 = load_a_frag(As[buf], 48, wm,      0, lane);
    v16bf a1 = load_a_frag(As[buf], 48, wm + 16, 0, lane);
#pragma unroll
    for (int m = 0; m < NMAT; ++m) {
      v16bf b0 = load_bt_frag(Wt[buf][m], 40, wn,      0, lane);
      v16bf b1 = load_bt_frag(Wt[buf][m], 40, wn + 16, 0, lane);
      acc[m][0][0] = wmma_bf16(a0, b0, acc[m][0][0]);
      acc[m][0][1] = wmma_bf16(a0, b1, acc[m][0][1]);
      acc[m][1][0] = wmma_bf16(a1, b0, acc[m][1][0]);
      acc[m][1][1] = wmma_bf16(a1, b1, acc[m][1][1]);
    }
  };

  stage(0, 0);
  async_fence();
  __syncthreads();
  int buf = 0;
  for (int k0 = 32; k0 < K; k0 += 32) {
    stage(buf ^ 1, k0);   // overlap next-tile staging with current compute
    compute(buf);
    async_fence();
    __syncthreads();
    buf ^= 1;
  }
  compute(buf);

#pragma unroll
  for (int m = 0; m < NMAT; ++m)
#pragma unroll
    for (int ti = 0; ti < 2; ++ti)
#pragma unroll
      for (int tj = 0; tj < 2; ++tj)
#pragma unroll
        for (int r = 0; r < 8; ++r) {
          int gm = m0 + wm + ti * 16 + r + hi * 8;
          int gn = n0 + wn + tj * 16 + (lane & 15);
          float v = acc[m][ti][tj][r] + bs[m][gn];
          Cs[m][(size_t)gm * N + gn] = (OutT)v;
        }
}

// ---------------- bucketed LSH attention ----------------
// One block per (b, h, bucket). Gathers 64 sorted tokens, 64x64 attention over
// dh=128, scatters output back to original token order (== unsort).
__global__ __launch_bounds__(256) void lsh_attention_kernel(const __bf16* __restrict__ qb,
                                                            const __bf16* __restrict__ vb,
                                                            const int* __restrict__ idx,
                                                            __bf16* __restrict__ ob) {
  __shared__ __align__(16) __bf16 Qs[BUCK * 136];   // [tok][d] stride 136
  __shared__ __align__(16) __bf16 Vt[DH * 72];      // transposed [d][tok] stride 72
  __shared__ float  Sc[BUCK * 68];                  // scores f32, stride 68
  __shared__ __align__(16) __bf16 Pm[BUCK * 72];    // probabilities, stride 72
  __shared__ int tok[BUCK];

  int g = blockIdx.x;
  int bucket = g & (NBUCK - 1);
  int h = (g >> 7) & (HEADS - 1);
  int b = g >> 10;
  int t = threadIdx.x;
  int lane = t & 31, wave = t >> 5;
  int hi = (lane >> 4) & 1;

  if (t < BUCK)
    tok[t] = idx[((size_t)b * HEADS + h) * SEQ + bucket * BUCK + t];
  __syncthreads();

  // gather: Q rows straight in (async b128), V rows transposed through VGPRs
  {
    int row = t >> 2, seg = t & 3;
    size_t src = ((size_t)b * SEQ + tok[row]) * DIM + h * DH + seg * 32;
#pragma unroll
    for (int u = 0; u < 4; ++u)
      copy_b128(qb + src + u * 8, Qs + row * 136 + seg * 32 + u * 8);
#pragma unroll
    for (int u = 0; u < 4; ++u) {
      v8bf vv = *(const v8bf*)(vb + src + u * 8);
#pragma unroll
      for (int j = 0; j < 8; ++j)
        Vt[(seg * 32 + u * 8 + j) * 72 + row] = vv[j];
    }
  }
  async_fence();
  __syncthreads();

  // scores = Q @ Q^T * scale  (16 tiles, 2 per wave, K=128)
  const float scale = 0.088388347648f;  // 1/sqrt(128)
#pragma unroll
  for (int tt = 0; tt < 2; ++tt) {
    int tile = wave * 2 + tt;
    int mi = tile >> 2, nj = tile & 3;
    v8f acc = (v8f)0.0f;
#pragma unroll
    for (int kk = 0; kk < DH; kk += 32) {
      v16bf a = load_a_frag(Qs, 136, mi * 16, kk, lane);
      v16bf bfrag = load_bt_frag(Qs, 136, nj * 16, kk, lane);  // B = Q^T
      acc = wmma_bf16(a, bfrag, acc);
    }
#pragma unroll
    for (int r = 0; r < 8; ++r)
      Sc[(mi * 16 + r + hi * 8) * 68 + nj * 16 + (lane & 15)] = acc[r] * scale;
  }
  __syncthreads();

  // softmax per row -> Pm (bf16, normalized)
  if (t < BUCK) {
    float mx = -3.4e38f;
    for (int j = 0; j < BUCK; ++j) mx = fmaxf(mx, Sc[t * 68 + j]);
    float sum = 0.f;
    float e[BUCK];
    for (int j = 0; j < BUCK; ++j) { e[j] = __expf(Sc[t * 68 + j] - mx); sum += e[j]; }
    float inv = 1.0f / sum;
    for (int j = 0; j < BUCK; ++j) Pm[t * 72 + j] = (__bf16)(e[j] * inv);
  }
  __syncthreads();

  // O = P @ V  (4x8 = 32 tiles, 4 per wave, K=64); B-fragments from Vt are
  // contiguous b128 reads. Scatter to original token order.
#pragma unroll
  for (int tt = 0; tt < 4; ++tt) {
    int tile = wave * 4 + tt;
    int mi = tile >> 3, nj = tile & 7;
    v8f acc = (v8f)0.0f;
#pragma unroll
    for (int kk = 0; kk < BUCK; kk += 32) {
      v16bf a = load_a_frag(Pm, 72, mi * 16, kk, lane);
      v16bf bfrag = load_bt_frag(Vt, 72, nj * 16, kk, lane);
      acc = wmma_bf16(a, bfrag, acc);
    }
#pragma unroll
    for (int r = 0; r < 8; ++r) {
      int m = mi * 16 + r + hi * 8;
      int n = nj * 16 + (lane & 15);
      ob[((size_t)b * SEQ + tok[m]) * DIM + h * DH + n] = (__bf16)acc[r];
    }
  }
}

// ---------------- launcher ----------------
extern "C" void kernel_launch(void* const* d_in, const int* in_sizes, int n_in,
                              void* d_out, int out_size, void* d_ws, size_t ws_size,
                              hipStream_t stream) {
  (void)in_sizes; (void)n_in; (void)out_size; (void)ws_size;
  const float* x     = (const float*)d_in[0];
  const float* Whash = (const float*)d_in[1];
  const float* Wq    = (const float*)d_in[2];
  const float* bq    = (const float*)d_in[3];
  const float* Wv    = (const float*)d_in[4];
  const float* bv    = (const float*)d_in[5];
  const float* Wo    = (const float*)d_in[6];
  const float* bo    = (const float*)d_in[7];
  float* out = (float*)d_out;

  char* ws = (char*)d_ws;
  size_t off = 0;
  auto alloc = [&](size_t bytes) -> char* {
    char* p = ws + off;
    off += (bytes + 255) & ~(size_t)255;
    return p;
  };
  __bf16* xb  = (__bf16*)alloc((size_t)TOK * DIM * 2);  // x bf16; reused as attn output
  __bf16* qb  = (__bf16*)alloc((size_t)TOK * DIM * 2);
  __bf16* vb  = (__bf16*)alloc((size_t)TOK * DIM * 2);
  __bf16* Wqb = (__bf16*)alloc((size_t)DIM * DIM * 2);
  __bf16* Wvb = (__bf16*)alloc((size_t)DIM * DIM * 2);
  __bf16* Wob = (__bf16*)alloc((size_t)DIM * DIM * 2);
  float*  ang = (float*) alloc((size_t)BATCH * HEADS * SEQ * 4);
  int*    idx = (int*)   alloc((size_t)BATCH * HEADS * SEQ * 4);

  const int nW = DIM * DIM;
  convert_bf16_kernel<<<(nW + 255) / 256, 256, 0, stream>>>(Wq, Wqb, nW);
  convert_bf16_kernel<<<(nW + 255) / 256, 256, 0, stream>>>(Wv, Wvb, nW);
  convert_bf16_kernel<<<(nW + 255) / 256, 256, 0, stream>>>(Wo, Wob, nW);

  hash_convert_kernel<<<TOK, 256, 0, stream>>>(x, Whash, ang, xb);
  bitonic_sort_kernel<<<BATCH * HEADS, 1024, 0, stream>>>(ang, idx);

  dim3 ggrid(DIM / 64, TOK / 128);
  // fused q & v projections: A tile staged once, 8 WMMAs per fragment pair
  gemm_multi_kernel<2, __bf16><<<ggrid, 256, 0, stream>>>(
      xb, Wqb, Wvb, bq, bv, qb, vb, TOK, DIM, DIM);

  lsh_attention_kernel<<<BATCH * HEADS * NBUCK, 256, 0, stream>>>(qb, vb, idx, xb /*reuse as ob*/);

  // output projection (f32 out)
  gemm_multi_kernel<1, float><<<ggrid, 256, 0, stream>>>(
      xb, Wob, Wob, bo, bo, out, out, TOK, DIM, DIM);
}
